// SpectralWormhole_56762287784343
// MI455X (gfx1250) — matrix operand order repaired and checked by reading, not compiled, not run on hardware
//
// SpectralWormhole on MI455X (gfx1250) — round 2.
// All WMMA operands are pre-converted bf16 and loaded as 128-bit vectors:
//   A fragment  = 2x global_load_b128 (rows of bf16 activations)
//   B fragment  = 2x global_load_b128 (weights pre-packed transposed [N][K] bf16,
//                 K^T from bf16 K rows, PV from transposed bf16 V [D][T])
// GEMM/attention epilogues stage tiles in LDS for coalesced b128 stores.
// Flash-style attention keeps all 12 [T,T] attention matrices out of HBM;
// directed attentions use the top-16 monotonicity trick so attn@V is a gather.
#include <hip/hip_runtime.h>
#include <math.h>

#define B_   8
#define T_   2048
#define D_   256
#define TOPK 16
#define LOG_T 7.6246189861593985f   // ln(2048)

typedef __attribute__((ext_vector_type(8)))  __bf16 v8bf;
typedef __attribute__((ext_vector_type(16))) __bf16 v16bf;
typedef __attribute__((ext_vector_type(8)))  float  v8f;

__device__ __forceinline__ v8f wmma_bf16(v16bf a, v16bf b, v8f c) {
  return __builtin_amdgcn_wmma_f32_16x16x32_bf16(false, a, false, b, (short)0, c, false, false);
}
__device__ __forceinline__ v16bf cat8(v8bf lo, v8bf hi) {
  return __builtin_shufflevector(lo, hi, 0, 1, 2, 3, 4, 5, 6, 7, 8, 9, 10, 11, 12, 13, 14, 15);
}
__device__ __forceinline__ v16bf zero16bf() {
  v16bf z;
#pragma unroll
  for (int i = 0; i < 16; ++i) z[i] = (__bf16)0.f;
  return z;
}

// ---------------------------------------------------------------- GEMM -----
// C[M,256] (+)= A[M,256] @ W + bias, optional per-row gate.  One wave = 16x64.
// A: bf16 row-major.  Wt: bf16 transposed [N][K].  Outputs: Cf (f32), Cb (bf16),
// Ct (bf16 transposed per-batch [D][T]) — any may be null.
__global__ void __launch_bounds__(32)
gemm_bf16_kernel(const __bf16* __restrict__ A, const __bf16* __restrict__ Wt,
                 const float* __restrict__ bias, const float* __restrict__ gate,
                 float* __restrict__ Cf, __bf16* __restrict__ Cb,
                 __bf16* __restrict__ Ct, int N, int K, int accumulate) {
  const int lane = threadIdx.x;
  const int row0 = blockIdx.x * 16;
  const int col0 = blockIdx.y * 64;

  v8f acc[4];
#pragma unroll
  for (int n = 0; n < 4; ++n)
#pragma unroll
    for (int r = 0; r < 8; ++r) acc[n][r] = 0.f;

  // hoisted per-lane base pointers -> loads become base + imm offsets
  const __bf16* pa = A + (size_t)(row0 + (lane & 15)) * K + ((lane >> 4) << 3);
  const __bf16* pw = Wt + (size_t)(col0 + (lane & 15)) * K + ((lane >> 4) << 3);

  for (int kc = 0; kc < K; kc += 32) {
    const v16bf a = cat8(*(const v8bf*)(pa + kc), *(const v8bf*)(pa + kc + 16));
#pragma unroll
    for (int n = 0; n < 4; ++n) {
      const __bf16* pwn = pw + (size_t)(16 * n) * K;
      const v16bf b = cat8(*(const v8bf*)(pwn + kc), *(const v8bf*)(pwn + kc + 16));
      acc[n] = wmma_bf16(a, b, acc[n]);
    }
  }

  __shared__ float tile[16][68];
  const int colc = lane & 15, rb = (lane >> 4) << 3;
#pragma unroll
  for (int n = 0; n < 4; ++n)
#pragma unroll
    for (int r = 0; r < 8; ++r) {
      float v = acc[n][r];
      if (bias) v += bias[col0 + 16 * n + colc];
      if (gate) v *= gate[row0 + rb + r];
      tile[rb + r][16 * n + colc] = v;
    }
  __syncthreads();

  if (Cf) {
#pragma unroll
    for (int it = 0; it < 8; ++it) {
      const int flat = it * 32 + lane;            // 256 float4 slots
      const int r = flat >> 4, c4 = (flat & 15) << 2;
      const float4 vv = *(const float4*)&tile[r][c4];
      float* dst = Cf + (size_t)(row0 + r) * N + col0 + c4;
      if (accumulate) { dst[0] += vv.x; dst[1] += vv.y; dst[2] += vv.z; dst[3] += vv.w; }
      else            *(float4*)dst = vv;
    }
  }
  if (Cb) {
#pragma unroll
    for (int it = 0; it < 4; ++it) {
      const int flat = it * 32 + lane;            // 128 v8bf slots
      const int r = flat >> 3, c8 = (flat & 7) << 3;
      v8bf o;
#pragma unroll
      for (int j = 0; j < 8; ++j) o[j] = (__bf16)tile[r][c8 + j];
      *(v8bf*)(Cb + (size_t)(row0 + r) * N + col0 + c8) = o;
    }
  }
  if (Ct) {                                       // Ct[b][col][t], rows are contiguous t
    const int r = lane & 15;
    const int bb = (row0 + r) >> 11;              // row0+r = b*T + t
    const int tt = (row0 + r) & (T_ - 1);
    __bf16* base = Ct + (size_t)bb * D_ * T_ + tt;
#pragma unroll
    for (int it = 0; it < 32; ++it) {
      const int c = it + ((lane >> 4) << 5);      // lanes 0-15: cols 0-31, 16-31: 32-63
      base[(size_t)(col0 + c) * T_] = (__bf16)tile[r][c];
    }
  }
}

// ------------------------------------------------- directed attention ------
// Causal scores via WMMA, online (m,S1,S2) entropy gate, top-16 sparsify,
// renormalized sparse AV gather.  One wave = 16 query rows.
__global__ void __launch_bounds__(32)
directed_attn_kernel(const __bf16* __restrict__ Q, const __bf16* __restrict__ Km,
                     const __bf16* __restrict__ V,
                     __bf16* __restrict__ attnout, float* __restrict__ gateval) {
  const int lane = threadIdx.x;
  const int b    = blockIdx.y;
  const int t0   = blockIdx.x * 16;
  const __bf16* Qb = Q + (size_t)b * T_ * D_;
  const __bf16* Kb = Km + (size_t)b * T_ * D_;
  const __bf16* Vb = V + (size_t)b * T_ * D_;

  __shared__ float sc[16][17];
  __shared__ float topv[16][TOPK];
  __shared__ int   topi[16][TOPK];
  __shared__ float wnorm[16][TOPK];

  const __bf16* pq = Qb + (size_t)(t0 + (lane & 15)) * D_ + ((lane >> 4) << 3);
  v16bf qf[8];
#pragma unroll
  for (int kc = 0; kc < 8; ++kc)
    qf[kc] = cat8(*(const v8bf*)(pq + kc * 32), *(const v8bf*)(pq + kc * 32 + 16));

  float m = -3.0e38f, S1 = 0.f, S2 = 0.f;
  if (lane < 16) {
#pragma unroll
    for (int j = 0; j < TOPK; ++j) { topv[lane][j] = -3.0e38f; topi[lane][j] = 0; }
  }
  __syncthreads();

  const __bf16* pk0 = Kb + (size_t)(lane & 15) * D_ + ((lane >> 4) << 3);
  const int smax = t0 + 16;
  for (int s0 = 0; s0 < smax; s0 += 16) {
    if (s0 + 16 < smax) __builtin_prefetch(Kb + (size_t)(s0 + 16) * D_, 0, 1);
    const __bf16* pk = pk0 + (size_t)s0 * D_;
    v8f c;
#pragma unroll
    for (int r = 0; r < 8; ++r) c[r] = 0.f;
#pragma unroll
    for (int kc = 0; kc < 8; ++kc) {
      const v16bf bf = cat8(*(const v8bf*)(pk + kc * 32), *(const v8bf*)(pk + kc * 32 + 16));
      c = wmma_bf16(qf[kc], bf, c);
    }
    {
      const int colc = lane & 15, rb = (lane >> 4) << 3;
#pragma unroll
      for (int r = 0; r < 8; ++r) sc[rb + r][colc] = c[r] * 0.0625f;  // 1/sqrt(256)
    }
    __syncthreads();
    if (lane < 16) {
      const int t = t0 + lane;
      for (int cc = 0; cc < 16; ++cc) {
        const int s = s0 + cc;
        if (s > t) break;                         // causal
        const float sv = sc[lane][cc];
        if (sv > m) {                             // online softmax rescale
          if (S1 > 0.f) {
            const float f = __expf(m - sv);
            S2 = (S2 + (m - sv) * S1) * f;
            S1 *= f;
          }
          m = sv;
        }
        const float e = __expf(sv - m);
        S1 += e; S2 += (sv - m) * e;
        int mi = 0; float mv = topv[lane][0];     // top-16 insert
#pragma unroll
        for (int j = 1; j < TOPK; ++j) {
          const float tv = topv[lane][j];
          if (tv < mv) { mv = tv; mi = j; }
        }
        if (sv > mv) { topv[lane][mi] = sv; topi[lane][mi] = s; }
      }
    }
    __syncthreads();
  }

  if (lane < 16) {
    const int t = t0 + lane;
    const float ent = __logf(S1) - S2 / S1;       // -sum p log p
    const float ne  = ent * (1.0f / LOG_T);
    const float g   = 1.f / (1.f + __expf(-(0.5f - ne) * 10.f));
    gateval[(size_t)b * T_ + t] = g;
    float Z = 0.f, w[TOPK];
#pragma unroll
    for (int j = 0; j < TOPK; ++j) {
      const float tv = topv[lane][j];
      const float e  = (tv > -1.0e37f) ? __expf(tv - m) : 0.f;
      w[j] = e; Z += e;
    }
    const float inv = 1.f / (Z + 1e-9f);
#pragma unroll
    for (int j = 0; j < TOPK; ++j) wnorm[lane][j] = w[j] * inv;
  }
  __syncthreads();

  // sparse AV: each lane owns 8 contiguous d -> one b128 V load per top-k entry
  for (int t = 0; t < 16; ++t) {
    float av[8];
#pragma unroll
    for (int e = 0; e < 8; ++e) av[e] = 0.f;
#pragma unroll
    for (int j = 0; j < TOPK; ++j) {
      const float wj = wnorm[t][j];
      const v8bf vv = *(const v8bf*)(Vb + (size_t)topi[t][j] * D_ + lane * 8);
#pragma unroll
      for (int e = 0; e < 8; ++e) av[e] += wj * (float)vv[e];
    }
    v8bf o;
#pragma unroll
    for (int e = 0; e < 8; ++e) o[e] = (__bf16)av[e];
    *(v8bf*)(attnout + ((size_t)b * T_ + t0 + t) * D_ + lane * 8) = o;
  }
}

// --------------------------------------------------- bridge attention ------
// Dense flash attention (scores + PV both WMMA); info = attn@V * gate / 6.
// Vt is the transposed bf16 V: Vt[b][d][t].
__global__ void __launch_bounds__(32)
bridge_attn_kernel(const __bf16* __restrict__ Q, const __bf16* __restrict__ Km,
                   const __bf16* __restrict__ Vt, __bf16* __restrict__ info) {
  const int lane = threadIdx.x;
  const int b    = blockIdx.y;
  const int t0   = blockIdx.x * 16;
  const __bf16* Qb  = Q + (size_t)b * T_ * D_;
  const __bf16* Kb  = Km + (size_t)b * T_ * D_;
  const __bf16* Vtb = Vt + (size_t)b * D_ * T_;

  __shared__ float  sc[16][17];
  __shared__ float  pb[16][16];
  __shared__ float  frow[16];
  __shared__ __bf16 ot[16][256];

  const __bf16* pq = Qb + (size_t)(t0 + (lane & 15)) * D_ + ((lane >> 4) << 3);
  v16bf qf[8];
#pragma unroll
  for (int kc = 0; kc < 8; ++kc)
    qf[kc] = cat8(*(const v8bf*)(pq + kc * 32), *(const v8bf*)(pq + kc * 32 + 16));

  v8f oacc[16];
#pragma unroll
  for (int n = 0; n < 16; ++n)
#pragma unroll
    for (int r = 0; r < 8; ++r) oacc[n][r] = 0.f;

  float m = -3.0e38f, S1 = 0.f, S2 = 0.f;
  const __bf16* pk0 = Kb + (size_t)(lane & 15) * D_ + ((lane >> 4) << 3);
  const __bf16* pv0 = Vtb + (size_t)(lane & 15) * T_;

  const int smax = t0 + 16;
  for (int s0 = 0; s0 < smax; s0 += 16) {
    if (s0 + 16 < smax) __builtin_prefetch(Kb + (size_t)(s0 + 16) * D_, 0, 1);
    const __bf16* pk = pk0 + (size_t)s0 * D_;
    v8f c;
#pragma unroll
    for (int r = 0; r < 8; ++r) c[r] = 0.f;
#pragma unroll
    for (int kc = 0; kc < 8; ++kc) {
      const v16bf bf = cat8(*(const v8bf*)(pk + kc * 32), *(const v8bf*)(pk + kc * 32 + 16));
      c = wmma_bf16(qf[kc], bf, c);
    }
    {
      const int colc = lane & 15, rb = (lane >> 4) << 3;
#pragma unroll
      for (int r = 0; r < 8; ++r) sc[rb + r][colc] = c[r] * 0.0625f;
    }
    __syncthreads();
    if (lane < 16) {
      const int t = t0 + lane;
      float mt = m;
#pragma unroll
      for (int cc = 0; cc < 16; ++cc)
        if (s0 + cc <= t) mt = fmaxf(mt, sc[lane][cc]);
      float f = 1.f;
      if (mt > m) {
        if (S1 > 0.f) {
          f = __expf(m - mt);
          S2 = (S2 + (m - mt) * S1) * f;
          S1 *= f;
        } else f = 0.f;
        m = mt;
      }
      frow[lane] = f;
#pragma unroll
      for (int cc = 0; cc < 16; ++cc) {
        float pv = 0.f;
        if (s0 + cc <= t) {
          const float sv = sc[lane][cc];
          pv = __expf(sv - m);
          S1 += pv; S2 += (sv - m) * pv;
        }
        pb[lane][cc] = pv;
      }
    }
    __syncthreads();
    {
      const int rb = (lane >> 4) << 3;            // rescale accumulators per row
#pragma unroll
      for (int n = 0; n < 16; ++n)
#pragma unroll
        for (int r = 0; r < 8; ++r) oacc[n][r] *= frow[rb + r];
    }
    v16bf pa;                                     // P tile as zero-padded 16x32 A frag
    {
      const int row = lane & 15;
      const int kb  = (lane >> 4) << 4;
#pragma unroll
      for (int r = 0; r < 8; ++r) {
        const int k = (r << 1) + kb;
        pa[2 * r]     = (k     < 16) ? (__bf16)pb[row][k]     : (__bf16)0.f;
        pa[2 * r + 1] = (k + 1 < 16) ? (__bf16)pb[row][k + 1] : (__bf16)0.f;
      }
    }
#pragma unroll
    for (int n = 0; n < 16; ++n) {
      v16bf vb;
      if (lane < 16) {
        const __bf16* pv = pv0 + (size_t)(16 * n) * T_ + s0;
        vb = cat8(*(const v8bf*)pv, *(const v8bf*)(pv + 8));
      } else {
        vb = zero16bf();                          // K 16..31 are zero-padded
      }
      oacc[n] = wmma_bf16(pa, vb, oacc[n]);
    }
    __syncthreads();
  }

  if (lane < 16) {
    const float ent = __logf(S1) - S2 / S1;
    const float ne  = ent * (1.0f / LOG_T);
    const float g   = 1.f / (1.f + __expf(-(0.5f - ne) * 10.f));
    frow[lane] = g * (1.0f / 6.0f) / S1;          // gate * 1/6 * softmax denom
  }
  __syncthreads();
  {
    const int colc = lane & 15, rb = (lane >> 4) << 3;
#pragma unroll
    for (int n = 0; n < 16; ++n)
#pragma unroll
      for (int r = 0; r < 8; ++r)
        ot[rb + r][16 * n + colc] = (__bf16)(oacc[n][r] * frow[rb + r]);
  }
  __syncthreads();
#pragma unroll
  for (int it = 0; it < 16; ++it) {               // coalesced b128 stores
    const int flat = it * 32 + lane;              // 512 v8bf slots
    const int r = flat >> 5, c8 = (flat & 31) << 3;
    *(v8bf*)(info + ((size_t)b * T_ + t0 + r) * D_ + c8) = *(const v8bf*)&ot[r][c8];
  }
}

// ------------------------------------------------------- elementwise -------
__global__ void gate_combine_kernel(const float4* __restrict__ band,
                                    const float4* __restrict__ glin,
                                    const float4* __restrict__ wh,
                                    float4* __restrict__ outp, int n4) {
  const int i = blockIdx.x * blockDim.x + threadIdx.x;
  if (i < n4) {
    const float4 gl = glin[i], bb = band[i], w = wh[i];
    float4 o;
    o.x = bb.x + w.x / (1.f + __expf(-gl.x));
    o.y = bb.y + w.y / (1.f + __expf(-gl.y));
    o.z = bb.z + w.z / (1.f + __expf(-gl.z));
    o.w = bb.w + w.w / (1.f + __expf(-gl.w));
    outp[i] = o;
  }
}

__global__ void f32_to_bf16_kernel(const float* __restrict__ src,
                                   __bf16* __restrict__ dst, int n8) {
  const int i = blockIdx.x * blockDim.x + threadIdx.x;
  if (i < n8) {
    const float4 a = ((const float4*)src)[2 * i];
    const float4 b = ((const float4*)src)[2 * i + 1];
    v8bf o;
    o[0] = (__bf16)a.x; o[1] = (__bf16)a.y; o[2] = (__bf16)a.z; o[3] = (__bf16)a.w;
    o[4] = (__bf16)b.x; o[5] = (__bf16)b.y; o[6] = (__bf16)b.z; o[7] = (__bf16)b.w;
    ((v8bf*)dst)[i] = o;
  }
}

// W[K,N] f32 row-major -> Wt[N,K] bf16 (WMMA B-operand friendly).
__global__ void pack_wt_kernel(const float* __restrict__ W, __bf16* __restrict__ Wt,
                               int K, int N) {
  const int nK8 = K >> 3;
  const int idx = blockIdx.x * blockDim.x + threadIdx.x;
  if (idx >= N * nK8) return;
  const int n = idx / nK8, k8 = (idx % nK8) << 3;
  v8bf o;
#pragma unroll
  for (int j = 0; j < 8; ++j) o[j] = (__bf16)W[(size_t)(k8 + j) * N + n];
  *(v8bf*)(Wt + (size_t)n * K + k8) = o;
}

// ----------------------------------------------------------- driver --------
extern "C" void kernel_launch(void* const* d_in, const int* in_sizes, int n_in,
                              void* d_out, int out_size, void* d_ws, size_t ws_size,
                              hipStream_t stream) {
  (void)in_sizes; (void)n_in; (void)out_size; (void)ws_size;
  const float* band[7];
  for (int i = 0; i < 7; ++i) band[i] = (const float*)d_in[i];
  const float* cWq = (const float*)d_in[7];  const float* cbq = (const float*)d_in[8];
  const float* cWk = (const float*)d_in[9];  const float* cbk = (const float*)d_in[10];
  const float* cWv = (const float*)d_in[11]; const float* cbv = (const float*)d_in[12];
  const float* cWo = (const float*)d_in[13]; const float* cbo = (const float*)d_in[14];
  const float* bWq = (const float*)d_in[15]; const float* bbq = (const float*)d_in[16];
  const float* bWk = (const float*)d_in[17]; const float* bbk = (const float*)d_in[18];
  const float* bWv = (const float*)d_in[19]; const float* bbv = (const float*)d_in[20];
  const float* bWo = (const float*)d_in[21]; const float* bbo = (const float*)d_in[22];
  const float* gW  = (const float*)d_in[23]; const float* gb  = (const float*)d_in[24];

  const size_t NE = (size_t)B_ * T_ * D_;
  const size_t DD = (size_t)D_ * D_;
  float* out[7];
  for (int i = 0; i < 7; ++i) out[i] = (float*)d_out + (size_t)i * NE;

  // workspace carve (256B aligned)
  size_t off = 0;
  char* base = (char*)d_ws;
  auto carve = [&](size_t bytes) -> void* {
    void* p = base + off;
    off += (bytes + 255) & ~(size_t)255;
    return p;
  };
  __bf16* bandb[7];
  for (int i = 0; i < 7; ++i) bandb[i] = (__bf16*)carve(NE * 2);
  __bf16* bufQ  = (__bf16*)carve(NE * 2);
  __bf16* bufK  = (__bf16*)carve(NE * 2);
  __bf16* bufVb = (__bf16*)carve(NE * 2);
  __bf16* bufVt = (__bf16*)carve(NE * 2);
  __bf16* bufAb = (__bf16*)carve(NE * 2);
  float*  bufWH  = (float*)carve(NE * 4);
  __bf16* bufWHb = (__bf16*)carve(NE * 2);
  float*  bufGL  = (float*)carve(NE * 4);
  float*  bufG   = (float*)carve((size_t)B_ * T_ * 4);
  __bf16* pWq = (__bf16*)carve(3 * DD * 2);
  __bf16* pWk = (__bf16*)carve(3 * DD * 2);
  __bf16* pWv = (__bf16*)carve(3 * DD * 2);
  __bf16* pWo = (__bf16*)carve(3 * DD * 2);
  __bf16* pbWq = (__bf16*)carve(DD * 2);
  __bf16* pbWk = (__bf16*)carve(DD * 2);
  __bf16* pbWv = (__bf16*)carve(DD * 2);
  __bf16* pbWo = (__bf16*)carve(DD * 2);
  __bf16* pgW = (__bf16*)carve(14 * DD * 2);     // [band][top/bottom][256][256]

  // ---- setup: convert bands, pack weights (bf16, transposed) ----
  {
    const int n8 = (int)(NE / 8);
    for (int i = 0; i < 7; ++i)
      f32_to_bf16_kernel<<<dim3((n8 + 255) / 256), dim3(256), 0, stream>>>(
          band[i], bandb[i], n8);
    const dim3 pk((D_ * D_ / 8 + 255) / 256), pkb(256);
    for (int p = 0; p < 3; ++p) {
      pack_wt_kernel<<<pk, pkb, 0, stream>>>(cWq + p * DD, pWq + p * DD, D_, D_);
      pack_wt_kernel<<<pk, pkb, 0, stream>>>(cWk + p * DD, pWk + p * DD, D_, D_);
      pack_wt_kernel<<<pk, pkb, 0, stream>>>(cWv + p * DD, pWv + p * DD, D_, D_);
      pack_wt_kernel<<<pk, pkb, 0, stream>>>(cWo + p * DD, pWo + p * DD, D_, D_);
    }
    pack_wt_kernel<<<pk, pkb, 0, stream>>>(bWq, pbWq, D_, D_);
    pack_wt_kernel<<<pk, pkb, 0, stream>>>(bWk, pbWk, D_, D_);
    pack_wt_kernel<<<pk, pkb, 0, stream>>>(bWv, pbWv, D_, D_);
    pack_wt_kernel<<<pk, pkb, 0, stream>>>(bWo, pbWo, D_, D_);
    for (int i = 0; i < 7; ++i) {
      pack_wt_kernel<<<pk, pkb, 0, stream>>>(gW + (size_t)i * 2 * DD,
                                             pgW + (size_t)(2 * i) * DD, D_, D_);
      pack_wt_kernel<<<pk, pkb, 0, stream>>>(gW + (size_t)i * 2 * DD + DD,
                                             pgW + (size_t)(2 * i + 1) * DD, D_, D_);
    }
  }

  const dim3 gGrid(B_ * T_ / 16, D_ / 64), gBlk(32);
  const dim3 aGrid(T_ / 16, B_), aBlk(32);
  auto gemm = [&](const __bf16* A, const __bf16* Wt, const float* bias,
                  const float* gate, float* Cf, __bf16* Cb, __bf16* Ct, int accum) {
    gemm_bf16_kernel<<<gGrid, gBlk, 0, stream>>>(A, Wt, bias, gate, Cf, Cb, Ct,
                                                 D_, D_, accum);
  };

  const int pairs[3][2] = {{0, 6}, {1, 5}, {2, 4}};
  for (int p = 0; p < 3; ++p) {
    for (int dir = 0; dir < 2; ++dir) {
      const int qs = dir == 0 ? pairs[p][0] : pairs[p][1];
      const int ks = dir == 0 ? pairs[p][1] : pairs[p][0];
      gemm(bandb[qs], pWq + p * DD, cbq + p * D_, nullptr, nullptr, bufQ, nullptr, 0);
      gemm(bandb[ks], pWk + p * DD, cbk + p * D_, nullptr, nullptr, bufK, nullptr, 0);
      gemm(bandb[ks], pWv + p * DD, cbv + p * D_, nullptr, nullptr, bufVb, nullptr, 0);
      directed_attn_kernel<<<aGrid, aBlk, 0, stream>>>(bufQ, bufK, bufVb, bufAb, bufG);
      // WH = (attnout @ Wo + bo) * gate   (f32 for combine, bf16 for gate GEMM)
      gemm(bufAb, pWo + p * DD, cbo + p * D_, bufG, bufWH, bufWHb, nullptr, 0);
      // glin = X @ Wg_top + b ; glin += WH @ Wg_bot   (concat split at K=256)
      gemm(bandb[qs], pgW + (size_t)(2 * qs) * DD, gb + qs * D_, nullptr,
           bufGL, nullptr, nullptr, 0);
      gemm(bufWHb, pgW + (size_t)(2 * qs + 1) * DD, nullptr, nullptr,
           bufGL, nullptr, nullptr, 1);
      gate_combine_kernel<<<dim3((unsigned)(NE / 4 + 255) / 256), dim3(256), 0, stream>>>(
          (const float4*)band[qs], (const float4*)bufGL, (const float4*)bufWH,
          (float4*)out[qs], (int)(NE / 4));
    }
  }

  // Bridge: outputs[3] = band3 + sum_i linear(info_i, Wo, bo)
  hipMemcpyAsync(out[3], band[3], NE * sizeof(float), hipMemcpyDeviceToDevice, stream);
  gemm(bandb[3], pbWq, bbq, nullptr, nullptr, bufQ, nullptr, 0);
  const int others[6] = {0, 1, 2, 4, 5, 6};
  for (int ii = 0; ii < 6; ++ii) {
    const int i = others[ii];
    gemm(bandb[i], pbWk, bbk, nullptr, nullptr, bufK, nullptr, 0);
    gemm(bandb[i], pbWv, bbv, nullptr, nullptr, nullptr, bufVt, 0);
    bridge_attn_kernel<<<aGrid, aBlk, 0, stream>>>(bufQ, bufK, bufVt, bufAb);
    gemm(bufAb, pbWo, bbo, nullptr, out[3], nullptr, nullptr, 1);  // accumulate
  }
}
